// FastSpeech2ConformerVarianceAdaptor_17678085390939
// MI455X (gfx1250) — matrix operand order repaired and checked by reading, not compiled, not run on hardware
//
#include <hip/hip_runtime.h>
#include <hip/hip_bf16.h>

// ---------------------------------------------------------------------------
// Problem constants (from reference)
// ---------------------------------------------------------------------------
#define BB      16
#define SS      512
#define DD      384          // D == H == 384
#define HH      384
#define KK      3
#define KDIM    (KK * DD)    // 1152, im2col inner dimension
#define ROWS    (BB * SS)    // 8192
#define MAX_OUT 4096         // S * MAX_DUR
#define NBINS   256

// GEMM tiling: block tile 128x128, staged K = 64 (two 32-wide WMMA microsteps)
#define GM  128
#define GN  128
#define GKS 64
#define NSTAGE (KDIM / GKS)  // 18

typedef __attribute__((ext_vector_type(16))) __bf16 v16bf;
typedef __attribute__((ext_vector_type(8)))  float  v8f;

// Exact parameter type of the async-LDS builtin (from hipcc diagnostic):
// generic pointer to a 16-byte int vector.
typedef int v4i __attribute__((__vector_size__(4 * sizeof(int))));

// ---------------------------------------------------------------------------
// Async global->LDS copy (CDNA5 GLOBAL_LOAD_ASYNC_TO_LDS_B128, ASYNCcnt path)
// with a synchronous fallback if the builtin is unavailable.
// ---------------------------------------------------------------------------
#if __has_builtin(__builtin_amdgcn_global_load_async_to_lds_b128) && \
    __has_builtin(__builtin_amdgcn_s_wait_asynccnt)
#define HAVE_ASYNC_LDS 1
#define ASYNC_CP16(dst, src)                                            \
    __builtin_amdgcn_global_load_async_to_lds_b128(                     \
        (v4i*)(src), (v4i*)(dst), 0, 0)
#define ASYNC_WAIT(n) __builtin_amdgcn_s_wait_asynccnt(n)
#else
#define HAVE_ASYNC_LDS 0
#define ASYNC_CP16(dst, src)                                            \
    (*reinterpret_cast<uint4*>(dst) = *reinterpret_cast<const uint4*>(src))
#define ASYNC_WAIT(n) ((void)0)
#endif

// ---------------------------------------------------------------------------
// bf16 helpers (storage = unsigned short)
// ---------------------------------------------------------------------------
__device__ __forceinline__ unsigned short f2bf(float f) {
    unsigned int u = __float_as_uint(f);
    unsigned int r = u + 0x7FFFu + ((u >> 16) & 1u);   // round-to-nearest-even
    return (unsigned short)(r >> 16);
}
__device__ __forceinline__ float bf2f(unsigned short h) {
    return __uint_as_float(((unsigned int)h) << 16);
}

// ---------------------------------------------------------------------------
// f32 -> bf16 cast
// ---------------------------------------------------------------------------
__global__ __launch_bounds__(256) void cast_f32_bf16(const float* __restrict__ x,
                                                     unsigned short* __restrict__ y,
                                                     int n) {
    int i = blockIdx.x * 256 + threadIdx.x;
    if (i < n) y[i] = f2bf(x[i]);
}

// 16B zero pad used as redirect target for out-of-range im2col rows
__global__ void init_zero16(uint4* z) {
    if (threadIdx.x == 0 && blockIdx.x == 0) *z = make_uint4(0u, 0u, 0u, 0u);
}

// ---------------------------------------------------------------------------
// Pack conv weights W[H, D, K] (f32, row-major) -> Wt[H, 3*D] bf16 where
// Wt[h, tap*D + d] = W[h, d, tap]  (K-transposed for contiguous LDS staging)
// ---------------------------------------------------------------------------
__global__ __launch_bounds__(256) void pack_weights(const float* __restrict__ W,
                                                    unsigned short* __restrict__ Wt) {
    int idx = blockIdx.x * 256 + threadIdx.x;       // [0, H*KDIM)
    if (idx >= HH * KDIM) return;
    int h   = idx / KDIM;
    int kk  = idx - h * KDIM;
    int tap = kk / DD;
    int d   = kk - tap * DD;
    Wt[idx] = f2bf(W[(h * DD + d) * KK + tap]);
}

// ---------------------------------------------------------------------------
// Conv-as-GEMM with WMMA bf16:  Y[p,h] = relu( bias[h] + sum_kk A[p,kk]*Wt[h,kk] )
// A = implicit im2col of X[B*S,384]: A[p, tap*384+d] = X[b, t+tap-1, d] (zero pad
// at sequence edges -> redirected to `zpad`). 384 % 64 == 0 so a 64-wide staged
// K chunk sits inside one tap: staging is plain (async) 16B copies.
// Block: 256 thr = 8 waves; wave tile 32x64 -> 2x4 accumulators; 8 wmma per
// 32-wide microstep, 16 per stage; 18 double-buffered async stages.
// ---------------------------------------------------------------------------
__global__ __launch_bounds__(256)
void conv_gemm_bf16(const unsigned short* __restrict__ X,   // [ROWS, 384] bf16
                    const unsigned short* __restrict__ Wt,  // [384, 1152] bf16
                    const float* __restrict__ bias,         // [384]
                    const unsigned short* __restrict__ zpad,// 16B of zeros
                    float* __restrict__ Y)                  // [ROWS, 384] f32
{
    __shared__ __align__(16) unsigned short Atile[2][GM * GKS];
    __shared__ __align__(16) unsigned short Btile[2][GN * GKS];

    const int tid   = threadIdx.x;
    const int lane  = tid & 31;
    const int wave  = tid >> 5;
    const int wm    = wave >> 1;          // 0..3
    const int wn    = wave & 1;           // 0..1
    const int Mbase = blockIdx.x * GM;
    const int Nbase = blockIdx.y * GN;
    const int l15   = lane & 15;
    const int hi16  = (lane >= 16) ? 1 : 0;

    v8f acc[2][4];
#pragma unroll
    for (int i = 0; i < 2; ++i)
#pragma unroll
        for (int j = 0; j < 4; ++j) acc[i][j] = (v8f)(0.0f);

    // ---- tile staging: 8 async b128 copies per thread per stage ------------
    auto stage = [&](int buf, int kt) {
        const int tap   = kt / DD;           // constant over the 64-chunk
        const int dbase = kt - tap * DD;
#pragma unroll
        for (int i = 0; i < 4; ++i) {
            int s   = tid + i * 256;         // 1024 segments of 8 bf16
            int row = s >> 3;                // 0..127
            int cs  = (s & 7) * 8;           // 0..56

            // A tile (im2col row; out-of-range -> zero pad redirect)
            int p  = Mbase + row;
            int b  = p >> 9;                 // /512
            int t  = p & (SS - 1);
            int tt = t + tap - 1;
            const unsigned short* asrc =
                (tt >= 0 && tt < SS)
                    ? X + ((size_t)((b << 9) + tt) * DD + dbase + cs)
                    : zpad;
            ASYNC_CP16(&Atile[buf][row * GKS + cs], asrc);

            // B tile (K-transposed weights, contiguous)
            const unsigned short* bsrc =
                Wt + ((size_t)(Nbase + row)) * KDIM + kt + cs;
            ASYNC_CP16(&Btile[buf][row * GKS + cs], bsrc);
        }
    };

    stage(0, 0);                             // prologue: fill buffer 0

    for (int it = 0; it < NSTAGE; ++it) {
        const int buf = it & 1;
        if (it + 1 < NSTAGE) {
            stage(buf ^ 1, (it + 1) * GKS);  // overlap next stage's DMA
            ASYNC_WAIT(8);                   // drain current stage (8 older ops)
        } else {
            ASYNC_WAIT(0);
        }
        __syncthreads();

        const unsigned int* Au = reinterpret_cast<const unsigned int*>(Atile[buf]);
        const unsigned int* Bu = reinterpret_cast<const unsigned int*>(Btile[buf]);

#pragma unroll
        for (int kc = 0; kc < GKS; kc += 32) {
            union Frag { v16bf v; unsigned int u[8]; };
            Frag afr[2], bfr[4];
#pragma unroll
            for (int f = 0; f < 2; ++f) {
                int m = wm * 32 + f * 16 + l15;
#pragma unroll
                for (int v = 0; v < 8; ++v) {
                    // A 16x32: lanes0-15 K={0..7,16..23}, lanes16-31 K={8..15,24..31}
                    int k0 = kc + ((v & 3) * 2) + ((v >> 2) * 16) + (hi16 ? 8 : 0);
                    afr[f].u[v] = Au[(m * GKS + k0) >> 1];
                }
            }
#pragma unroll
            for (int f = 0; f < 4; ++f) {
                int n = wn * 64 + f * 16 + l15;
#pragma unroll
                for (int v = 0; v < 8; ++v) {
                    // B 32x16: lanes0-15 K={0..7,16..23}, lanes16-31 K={8..15,24..31}
                    int k0 = kc + ((v & 3) * 2) + ((v >> 2) * 16) + (hi16 ? 8 : 0);
                    bfr[f].u[v] = Bu[(n * GKS + k0) >> 1];
                }
            }
#pragma unroll
            for (int fm = 0; fm < 2; ++fm)
#pragma unroll
                for (int fn = 0; fn < 4; ++fn)
                    acc[fm][fn] = __builtin_amdgcn_wmma_f32_16x16x32_bf16(
                        false, afr[fm].v, false, bfr[fn].v,
                        (short)0, acc[fm][fn], false, false);
        }
        __syncthreads();                     // all waves done reading buf
    }

    // ---- epilogue: bias + ReLU, f32 store (C/D: lane->n, vgpr->m) ----------
#pragma unroll
    for (int fm = 0; fm < 2; ++fm) {
#pragma unroll
        for (int fn = 0; fn < 4; ++fn) {
            int ncol = Nbase + wn * 64 + fn * 16 + l15;
            float bv = bias[ncol];
#pragma unroll
            for (int v = 0; v < 8; ++v) {
                int m   = Mbase + wm * 32 + fm * 16 + (hi16 ? 8 : 0) + v;
                float y = acc[fm][fn][v] + bv;
                Y[(size_t)m * HH + ncol] = (y > 0.0f) ? y : 0.0f;
            }
        }
    }
}

// ---------------------------------------------------------------------------
// LayerNorm over last dim (384), output bf16. One block (128 thr) per row.
// ---------------------------------------------------------------------------
__global__ __launch_bounds__(128)
void layernorm_bf16(const float* __restrict__ X, const float* __restrict__ g,
                    const float* __restrict__ beta, unsigned short* __restrict__ Y)
{
    __shared__ float rs[128], rq[128];
    const int tid = threadIdx.x;
    const float* xr = X + (size_t)blockIdx.x * HH;
    float v0 = xr[tid], v1 = xr[tid + 128], v2 = xr[tid + 256];
    rs[tid] = v0 + v1 + v2;
    rq[tid] = v0 * v0 + v1 * v1 + v2 * v2;
    __syncthreads();
    for (int off = 64; off > 0; off >>= 1) {
        if (tid < off) { rs[tid] += rs[tid + off]; rq[tid] += rq[tid + off]; }
        __syncthreads();
    }
    float mean = rs[0] * (1.0f / HH);
    float var  = rq[0] * (1.0f / HH) - mean * mean;
    float rstd = rsqrtf(var + 1e-5f);
    unsigned short* yr = Y + (size_t)blockIdx.x * HH;
    yr[tid]       = f2bf((v0 - mean) * rstd * g[tid]       + beta[tid]);
    yr[tid + 128] = f2bf((v1 - mean) * rstd * g[tid + 128] + beta[tid + 128]);
    yr[tid + 256] = f2bf((v2 - mean) * rstd * g[tid + 256] + beta[tid + 256]);
}

// ---------------------------------------------------------------------------
// Projection H->1: one wave per row, wave32 shuffle reduction.
// ---------------------------------------------------------------------------
__global__ __launch_bounds__(256)
void project(const unsigned short* __restrict__ X, const float* __restrict__ wp,
             const float* __restrict__ bp, float* __restrict__ out)
{
    int lane = threadIdx.x & 31;
    int row  = blockIdx.x * 8 + (threadIdx.x >> 5);
    const unsigned short* xr = X + (size_t)row * HH;
    float acc = 0.0f;
#pragma unroll
    for (int i = 0; i < 12; ++i) {
        int e = lane + i * 32;
        acc += bf2f(xr[e]) * wp[e];
    }
#pragma unroll
    for (int off = 16; off > 0; off >>= 1) acc += __shfl_down(acc, off, 32);
    if (lane == 0) out[row] = acc + bp[0];
}

// ---------------------------------------------------------------------------
// hidden += table[searchsorted(linspace(-1,1,255), s, left)]; optional bf16 copy.
// ---------------------------------------------------------------------------
__global__ __launch_bounds__(128)
void add_embed(const float* __restrict__ Xin, const float* __restrict__ scalar,
               const float* __restrict__ table, float* __restrict__ Xout,
               unsigned short* __restrict__ Ybf)
{
    int r = blockIdx.x;
    float v = scalar[r];
    // left searchsorted on linspace(-1,1,255): i = ceil((v+1)/step), step = 2/254
    int idx = (int)ceilf((v + 1.0f) * 127.0f);
    idx = (idx < 0) ? 0 : ((idx > NBINS - 1) ? NBINS - 1 : idx);
    const float* tr = table + (size_t)idx * DD;
    const float* xr = Xin + (size_t)r * DD;
    float* yr = Xout + (size_t)r * DD;
#pragma unroll
    for (int i = 0; i < 3; ++i) {
        int e = threadIdx.x + i * 128;
        float s = xr[e] + tr[e];
        yr[e] = s;
        if (Ybf) Ybf[(size_t)r * DD + e] = f2bf(s);
    }
}

// ---------------------------------------------------------------------------
// Inclusive scan of durations (512 per batch row) + lengths output.
// ---------------------------------------------------------------------------
__global__ __launch_bounds__(512)
void scan_durations(const int* __restrict__ dur, int* __restrict__ cum,
                    float* __restrict__ lengths)
{
    __shared__ int buf[SS];
    int b = blockIdx.x, tid = threadIdx.x;
    buf[tid] = dur[b * SS + tid];
    __syncthreads();
    for (int off = 1; off < SS; off <<= 1) {
        int add = (tid >= off) ? buf[tid - off] : 0;
        __syncthreads();
        buf[tid] += add;
        __syncthreads();
    }
    cum[b * SS + tid] = buf[tid];
    if (tid == SS - 1) lengths[b] = (float)buf[SS - 1];
}

// ---------------------------------------------------------------------------
// Length regulation: out[b,pos,:] = hid[b, min(searchsorted_right(cum,pos),511), :]
// masked by pos < len. One block (128 thr) per (pos, b).
// ---------------------------------------------------------------------------
__global__ __launch_bounds__(128)
void length_regulate(const float* __restrict__ Hid, const int* __restrict__ cum,
                     float* __restrict__ out)
{
    int b = blockIdx.y, pos = blockIdx.x;
    const int* c = cum + b * SS;
    int lo = 0, hicnt = SS;
    while (lo < hicnt) {                   // first i with c[i] > pos
        int mid = (lo + hicnt) >> 1;
        if (c[mid] > pos) hicnt = mid; else lo = mid + 1;
    }
    int idx   = (lo > SS - 1) ? SS - 1 : lo;
    float msk = (pos < c[SS - 1]) ? 1.0f : 0.0f;
    const float* src = Hid + ((size_t)(b * SS + idx)) * DD;
    float*       dst = out + ((size_t)(b * MAX_OUT + pos)) * DD;
    for (int i = threadIdx.x; i < DD; i += 128) dst[i] = src[i] * msk;
}

// ---------------------------------------------------------------------------
// Host-side orchestration
// ---------------------------------------------------------------------------
static void run_predictor(const unsigned short* Xbf, void* const* d_in, int P,
                          unsigned short* Wt, float* act, unsigned short* Hbf,
                          const unsigned short* zpad, float* out_scalar,
                          hipStream_t stream)
{
    const float* w1  = (const float*)d_in[P + 0];
    const float* b1  = (const float*)d_in[P + 1];
    const float* g1  = (const float*)d_in[P + 2];
    const float* be1 = (const float*)d_in[P + 3];
    const float* w2  = (const float*)d_in[P + 4];
    const float* b2  = (const float*)d_in[P + 5];
    const float* g2  = (const float*)d_in[P + 6];
    const float* be2 = (const float*)d_in[P + 7];
    const float* wp  = (const float*)d_in[P + 8];
    const float* bp  = (const float*)d_in[P + 9];

    dim3 gemm_grid(ROWS / GM, HH / GN);    // 64 x 3

    pack_weights<<<(HH * KDIM) / 256, 256, 0, stream>>>(w1, Wt);
    conv_gemm_bf16<<<gemm_grid, 256, 0, stream>>>(Xbf, Wt, b1, zpad, act);
    layernorm_bf16<<<ROWS, 128, 0, stream>>>(act, g1, be1, Hbf);
    pack_weights<<<(HH * KDIM) / 256, 256, 0, stream>>>(w2, Wt);
    conv_gemm_bf16<<<gemm_grid, 256, 0, stream>>>(Hbf, Wt, b2, zpad, act);
    layernorm_bf16<<<ROWS, 128, 0, stream>>>(act, g2, be2, Hbf);
    project<<<ROWS / 8, 256, 0, stream>>>(Hbf, wp, bp, out_scalar);
}

extern "C" void kernel_launch(void* const* d_in, const int* in_sizes, int n_in,
                              void* d_out, int out_size, void* d_ws, size_t ws_size,
                              hipStream_t stream)
{
    (void)in_sizes; (void)n_in; (void)out_size; (void)ws_size;

    const float* hidden       = (const float*)d_in[0];
    const int*   durations    = (const int*)d_in[2];
    const float* embed_pitch  = (const float*)d_in[33];
    const float* embed_energy = (const float*)d_in[34];

    // workspace layout (bytes)
    char* ws = (char*)d_ws;
    const size_t XBF_B = (size_t)ROWS * DD * 2;       // 6,291,456
    const size_t ACT_B = (size_t)ROWS * DD * 4;       // 12,582,912
    const size_t WT_B  = (size_t)HH * KDIM * 2;       // 884,736
    unsigned short* Xbf  = (unsigned short*)(ws);
    unsigned short* X2bf = (unsigned short*)(ws + XBF_B);
    unsigned short* Hbf  = (unsigned short*)(ws + 2 * XBF_B);
    float*          act  = (float*)(ws + 3 * XBF_B);
    unsigned short* Wt   = (unsigned short*)(ws + 3 * XBF_B + ACT_B);
    float*          hid2 = (float*)(ws + 3 * XBF_B + ACT_B + ((WT_B + 255) & ~255ULL));
    int*            cum  = (int*)((char*)hid2 + ACT_B);
    uint4*          zpad = (uint4*)((char*)cum + (((size_t)BB * SS * 4 + 255) & ~255ULL));

    // output layout (concatenated, f32)
    float* out     = (float*)d_out;
    float* lengths = out + (size_t)BB * MAX_OUT * DD;
    float* logd    = lengths + BB;
    float* pitch   = logd + ROWS;
    float* energy  = pitch + ROWS;

    // 0) zero pad block for im2col edge redirect
    init_zero16<<<1, 32, 0, stream>>>(zpad);

    // 1) bf16 activations of input hidden
    cast_f32_bf16<<<(ROWS * DD) / 256, 256, 0, stream>>>(hidden, Xbf, ROWS * DD);

    // 2) duration predictor -> log_duration_out
    run_predictor(Xbf, d_in, 3, Wt, act, Hbf, (const unsigned short*)zpad, logd, stream);

    // 3) pitch predictor -> pitch_out ; hidden2 = hidden + embed_pitch[idx]
    run_predictor(Xbf, d_in, 13, Wt, act, Hbf, (const unsigned short*)zpad, pitch, stream);
    add_embed<<<ROWS, 128, 0, stream>>>(hidden, pitch, embed_pitch, hid2, X2bf);

    // 4) energy predictor on hidden2 -> energy_out ; hidden3 (in place)
    run_predictor(X2bf, d_in, 23, Wt, act, Hbf, (const unsigned short*)zpad, energy, stream);
    add_embed<<<ROWS, 128, 0, stream>>>(hid2, energy, embed_energy, hid2,
                                        (unsigned short*)nullptr);

    // 5) length regulation
    scan_durations<<<BB, SS, 0, stream>>>(durations, cum, lengths);
    length_regulate<<<dim3(MAX_OUT, BB), 128, 0, stream>>>(hid2, cum, out);
}
